// MemoryAttention_46943992545665
// MI455X (gfx1250) — matrix operand order repaired and checked by reading, not compiled, not run on hardware
//
#include <hip/hip_runtime.h>
#include <hip/hip_bf16.h>

// ---------------------------------------------------------------------------
// MemoryAttention for MI455X (gfx1250, wave32).
// bf16 WMMA (v_wmma_f32_16x16x32_bf16) for all matmuls, f32 accumulate,
// flash-attention with online softmax; rolled-KV concat handled by index map.
// Global->LDS tile staging uses GLOBAL_LOAD_ASYNC_TO_LDS_B128 when available.
// ---------------------------------------------------------------------------

typedef __attribute__((ext_vector_type(16))) __bf16 v16bf;
typedef __attribute__((ext_vector_type(8)))  __bf16 v8bf;
typedef __attribute__((ext_vector_type(8)))  float  v8f;
typedef __attribute__((__vector_size__(4 * sizeof(int)))) int v4i;

union BF16x16 { v16bf v; v8bf h[2]; };

#define AS1 __attribute__((address_space(1)))
#define AS3 __attribute__((address_space(3)))

#if defined(__has_builtin)
#if __has_builtin(__builtin_amdgcn_global_load_async_to_lds_b128) && \
    __has_builtin(__builtin_amdgcn_s_wait_asynccnt)
#define USE_ASYNC_LDS 1
#endif
#endif
#ifndef USE_ASYNC_LDS
#define USE_ASYNC_LDS 0
#endif

// 16-byte global -> LDS copy (async on gfx1250: GLOBAL_LOAD_ASYNC_TO_LDS_B128)
__device__ __forceinline__ void copy16_g2l(const __bf16* g, __bf16* l) {
#if USE_ASYNC_LDS
    __builtin_amdgcn_global_load_async_to_lds_b128(
        (AS1 v4i*)(g), (AS3 v4i*)(l), /*offset=*/0, /*cpol=*/0);
#else
    *(v8bf*)l = *(const v8bf*)g;
#endif
}

// Complete all outstanding async LDS copies (no-op on fallback path).
__device__ __forceinline__ void async_fence() {
#if USE_ASYNC_LDS
    __builtin_amdgcn_s_wait_asynccnt(0);
#endif
}

// ---- fragment loaders (CDNA5 ISA sec 7.12.2, wave32 16-bit layouts) ----
// A (16xK=32): lanes 0-15 hold M=lane, elems 0-7 = k 0..7, elems 8-15 = k 16..23
//              lanes 16-31 hold M=lane-16, k 8..15 and k 24..31.
__device__ __forceinline__ v16bf load_a_frag(const __bf16* rowptr, int laneHalf) {
    BF16x16 f;
    const v8bf* p = (const v8bf*)rowptr;
    f.h[0] = p[laneHalf ? 1 : 0];
    f.h[1] = p[laneHalf ? 3 : 2];
    return f.v;
}
// B (K=32 x 16), stored as B^T rows (N-major, k contiguous):
// lanes 0-15: N=lane, k 0..7 and 16..23 ; lanes 16-31: N=lane-16, k 8..15 and 24..31.
__device__ __forceinline__ v16bf load_b_frag(const __bf16* rowptr, int laneHalf) {
    BF16x16 f;
    const v8bf* p = (const v8bf*)rowptr;
    f.h[0] = p[laneHalf ? 1 : 0];
    f.h[1] = p[laneHalf ? 3 : 2];
    return f.v;
}

__device__ __forceinline__ v8f wmma_bf16(v16bf a, v16bf b, v8f c) {
    return __builtin_amdgcn_wmma_f32_16x16x32_bf16(
        /*neg_a=*/false, a, /*neg_b=*/false, b,
        /*c_mod=*/(short)0, c, /*reuse_a=*/false, /*reuse_b=*/false);
}

// Map extended-KV row s in [0,4608) of batch b -> (src_batch, src_time).
// Reference: sk = stack([roll(k[:, :512], i, axis=0) for i in 0..6]).reshape(B,-1,C)
//            kext = concat(sk, k)  -> S = 7*512 + 1024 = 4608
__device__ __forceinline__ void map_s(int b, int s, int& sb, int& st) {
    if (s >= 3584) { sb = b; st = s - 3584; }
    else {
        int g  = b * 3584 + s;      // global row over (7, 8, 512)
        int i  = g >> 12;           // / 4096
        int r  = g & 4095;
        int b2 = r >> 9;            // / 512
        st     = r & 511;
        sb     = (b2 - i) & 7;      // roll along batch
    }
}

// ---------------------------------------------------------------------------
// elementwise convert kernels
// ---------------------------------------------------------------------------
__global__ __launch_bounds__(256) void conv_scale_kernel(const float* __restrict__ in,
                                                         __bf16* __restrict__ out,
                                                         int n, float scale) {
    for (int i = blockIdx.x * 256 + threadIdx.x; i < n; i += gridDim.x * 256)
        out[i] = (__bf16)(in[i] * scale);
}

// Wt[n*K + k] = bf16( W[k*N + n] )   (W is K x N row-major)
__global__ __launch_bounds__(256) void transpose_conv_kernel(const float* __restrict__ W,
                                                             __bf16* __restrict__ Wt,
                                                             int K, int N) {
    int total = K * N;
    for (int i = blockIdx.x * 256 + threadIdx.x; i < total; i += gridDim.x * 256) {
        int n = i / K, k = i - n * K;
        Wt[i] = (__bf16)W[(size_t)k * N + n];
    }
}

// ---------------------------------------------------------------------------
// Generic bf16 WMMA GEMM:  C[M,N] = A[M,K] * Bt[N,K]^T
// Block: 256 threads (8 waves), tile 128(M) x 64(N). K multiple of 32.
// Cb!=null -> bf16 output, else Cf f32 output.
// ---------------------------------------------------------------------------
__global__ __launch_bounds__(256) void gemm_wmma_kernel(const __bf16* __restrict__ A,
                                                        const __bf16* __restrict__ Bt,
                                                        __bf16* __restrict__ Cb,
                                                        float* __restrict__ Cf,
                                                        int M, int N, int K) {
    __shared__ __bf16 As[128][32];
    __shared__ __bf16 Bts[64][32];
    const int m0 = blockIdx.x * 128;
    const int n0 = blockIdx.y * 64;
    const int t  = threadIdx.x;
    const int lane = t & 31, w = t >> 5;
    const int laneHalf = (lane >= 16);
    const int l15 = lane & 15;

    v8f acc[4] = {};

    for (int k0 = 0; k0 < K; k0 += 32) {
        // A tile: 128 rows x 32 cols = 512 chunks of 8 bf16 (2 per thread)
        {
            int row0 = t >> 2, cc0 = t & 3;                  // chunks t, t+256
            copy16_g2l(A + (size_t)(m0 + row0) * K + k0 + cc0 * 8, &As[row0][cc0 * 8]);
            int row1 = (t + 256) >> 2, cc1 = t & 3;
            copy16_g2l(A + (size_t)(m0 + row1) * K + k0 + cc1 * 8, &As[row1][cc1 * 8]);
        }
        // B^T tile: 64 rows x 32 cols = 256 chunks (1 per thread)
        {
            int row = t >> 2, cc = t & 3;
            copy16_g2l(Bt + (size_t)(n0 + row) * K + k0 + cc * 8, &Bts[row][cc * 8]);
        }
        async_fence();
        __syncthreads();

        v16bf a = load_a_frag(&As[w * 16 + l15][0], laneHalf);
        #pragma unroll
        for (int nf = 0; nf < 4; ++nf) {
            v16bf bfr = load_b_frag(&Bts[nf * 16 + l15][0], laneHalf);
            acc[nf] = wmma_bf16(a, bfr, acc[nf]);
        }
        __syncthreads();
    }

    // C layout: VGPR r -> row (laneHalf? 8+r : r), col = lane&15 within 16x16 tile
    #pragma unroll
    for (int nf = 0; nf < 4; ++nf) {
        #pragma unroll
        for (int r = 0; r < 8; ++r) {
            int row = m0 + w * 16 + (laneHalf ? 8 + r : r);
            int col = n0 + nf * 16 + l15;
            if (Cf) Cf[(size_t)row * N + col] = acc[nf][r];
            else    Cb[(size_t)row * N + col] = (__bf16)acc[nf][r];
        }
    }
}

// ---------------------------------------------------------------------------
// Flash attention over the extended (rolled+concat) KV.
// Grid: (B*H, T/128). Block: 256 threads = 8 waves; wave w owns q-rows w*16..+16.
// qb: [B,T,C] bf16 (pre-scaled by 1/sqrt(D));  kvb: [B,T,2C] bf16 (k | v)
// yb: [B,T,C] bf16 output (pre-projection).
// ---------------------------------------------------------------------------
__global__ __launch_bounds__(256) void attn_wmma_kernel(const __bf16* __restrict__ qb,
                                                        const __bf16* __restrict__ kvb,
                                                        __bf16* __restrict__ yb) {
    __shared__ __bf16 Qs[128][64];     // 16 KB
    __shared__ __bf16 Ks[32][64];      //  4 KB  (keys x D, == B^T for QK^T)
    __shared__ __bf16 Vt[64][32];      //  4 KB  (D x keys, == B^T for PV)
    __shared__ __bf16 Pw[8][16][32];   //  8 KB  per-wave P staging

    const int bh = blockIdx.x;
    const int b = bh >> 3, h = bh & 7;
    const int t0 = blockIdx.y * 128;
    const int t  = threadIdx.x;
    const int lane = t & 31, w = t >> 5;
    const int laneHalf = (lane >= 16);
    const int l15 = lane & 15;

    // Load Q tile (128 x 64): 1024 chunks of 8 bf16, async staged
    #pragma unroll
    for (int c = t; c < 1024; c += 256) {
        int row = c >> 3, dstart = (c & 7) * 8;
        copy16_g2l(qb + (size_t)(b * 1024 + t0 + row) * 512 + h * 64 + dstart,
                   &Qs[row][dstart]);
    }

    float m[8], lsum[8];
    v8f acc[4] = {};
    #pragma unroll
    for (int r = 0; r < 8; ++r) { m[r] = -3.0e38f; lsum[r] = 0.0f; }

    const int key = t >> 3;               // this thread's KV-tile row
    const int dstart = (t & 7) * 8;       // and 8-element d-chunk

    for (int s0 = 0; s0 < 4608; s0 += 32) {
        __syncthreads();
        // ---- load K tile (32x64, async) and V^T tile (64x32) via roll/concat map ----
        {
            int sb, st; map_s(b, s0 + key, sb, st);
            size_t base = (size_t)(sb * 1024 + st) * 1024 + h * 64 + dstart;
            copy16_g2l(kvb + base, &Ks[key][dstart]);          // K: direct 16B chunks
            v8bf vv = *(const v8bf*)(kvb + base + 512);        // V: transpose in regs
            #pragma unroll
            for (int j = 0; j < 8; ++j) Vt[dstart + j][key] = vv[j];
            if (s0 + 32 < 4608) {  // prefetch next tile -> global_prefetch_b8
                int sb2, st2; map_s(b, s0 + 32 + key, sb2, st2);
                __builtin_prefetch(kvb + (size_t)(sb2 * 1024 + st2) * 1024 + h * 64 + dstart, 0, 1);
            }
        }
        async_fence();
        __syncthreads();

        // ---- S = Q * K^T (16 q-rows x 32 keys), contraction over D=64 ----
        v8f sc0 = {}, sc1 = {};
        #pragma unroll
        for (int c = 0; c < 2; ++c) {
            v16bf a  = load_a_frag(&Qs[w * 16 + l15][c * 32], laneHalf);
            v16bf b0 = load_b_frag(&Ks[l15][c * 32], laneHalf);
            v16bf b1 = load_b_frag(&Ks[16 + l15][c * 32], laneHalf);
            sc0 = wmma_bf16(a, b0, sc0);
            sc1 = wmma_bf16(a, b1, sc1);
        }

        // ---- online softmax (row stats live in the 16-lane half owning the row) ----
        #pragma unroll
        for (int r = 0; r < 8; ++r) {
            float mx = fmaxf(sc0[r], sc1[r]);
            #pragma unroll
            for (int off = 8; off; off >>= 1) mx = fmaxf(mx, __shfl_xor(mx, off, 32));
            float mnew = fmaxf(m[r], mx);
            float corr = __expf(m[r] - mnew);
            float p0 = __expf(sc0[r] - mnew);
            float p1 = __expf(sc1[r] - mnew);
            float rs = p0 + p1;
            #pragma unroll
            for (int off = 8; off; off >>= 1) rs += __shfl_xor(rs, off, 32);
            lsum[r] = lsum[r] * corr + rs;
            m[r] = mnew;
            #pragma unroll
            for (int nf = 0; nf < 4; ++nf) acc[nf][r] *= corr;
            int rr = laneHalf ? 8 + r : r;   // C-layout row owned by this lane
            Pw[w][rr][l15]      = (__bf16)p0;
            Pw[w][rr][16 + l15] = (__bf16)p1;
        }

        // ---- acc += P * V  (contraction over 32 keys; per-wave LDS, DS in-order) ----
        v16bf pa = load_a_frag(&Pw[w][l15][0], laneHalf);
        #pragma unroll
        for (int nf = 0; nf < 4; ++nf) {
            v16bf bv = load_b_frag(&Vt[nf * 16 + l15][0], laneHalf);
            acc[nf] = wmma_bf16(pa, bv, acc[nf]);
        }
    }

    // ---- normalize and store y (bf16, pre-projection) ----
    #pragma unroll
    for (int r = 0; r < 8; ++r) {
        float inv = 1.0f / lsum[r];
        int row = t0 + w * 16 + (laneHalf ? 8 + r : r);
        #pragma unroll
        for (int nf = 0; nf < 4; ++nf)
            yb[(size_t)(b * 1024 + row) * 512 + h * 64 + nf * 16 + l15] =
                (__bf16)(acc[nf][r] * inv);
    }
}

// ---------------------------------------------------------------------------
// launch
// ---------------------------------------------------------------------------
extern "C" void kernel_launch(void* const* d_in, const int* in_sizes, int n_in,
                              void* d_out, int out_size, void* d_ws, size_t ws_size,
                              hipStream_t stream) {
    const float* x     = (const float*)d_in[0];   // (8,1024,512)
    const float* q     = (const float*)d_in[1];   // (8,1024,512)
    const float* Wkv   = (const float*)d_in[2];   // (512,1024)
    const float* Wproj = (const float*)d_in[3];   // (512,512)
    float* out = (float*)d_out;                   // (8,1024,512) f32

    char* ws = (char*)d_ws;
    __bf16* xb     = (__bf16*)(ws);                 //  8 MB: x bf16
    __bf16* qb     = (__bf16*)(ws + 8388608);       //  8 MB: q*scale bf16
    __bf16* kvb    = (__bf16*)(ws + 16777216);      // 16 MB: kv = x@W_kv bf16
    __bf16* yb     = (__bf16*)(ws + 33554432);      //  8 MB: attention out bf16
    __bf16* WkvT   = (__bf16*)(ws + 41943040);      //  1 MB: W_kv^T bf16 (1024x512)
    __bf16* WprojT = (__bf16*)(ws + 42991616);      // 512KB: W_proj^T bf16 (512x512)

    const float scale = 0.125f;  // 1/sqrt(D=64), folded into Q

    conv_scale_kernel<<<1024, 256, 0, stream>>>(x, xb, 8 * 1024 * 512, 1.0f);
    conv_scale_kernel<<<1024, 256, 0, stream>>>(q, qb, 8 * 1024 * 512, scale);
    transpose_conv_kernel<<<512, 256, 0, stream>>>(Wkv,   WkvT,   512, 1024);
    transpose_conv_kernel<<<256, 256, 0, stream>>>(Wproj, WprojT, 512, 512);

    // kv = x @ W_kv : (8192 x 512) * (512 x 1024) -> bf16
    dim3 g1(8192 / 128, 1024 / 64);
    gemm_wmma_kernel<<<g1, 256, 0, stream>>>(xb, WkvT, kvb, nullptr, 8192, 1024, 512);

    // flash attention over extended KV (S = 4608)
    dim3 g2(8 * 8, 1024 / 128);
    attn_wmma_kernel<<<g2, 256, 0, stream>>>(qb, kvb, yb);

    // out = y @ W_proj : (8192 x 512) * (512 x 512) -> f32
    dim3 g3(8192 / 128, 512 / 64);
    gemm_wmma_kernel<<<g3, 256, 0, stream>>>(yb, WprojT, nullptr, out, 8192, 512, 512);
}
